// CFGATLayer_84653805404337
// MI455X (gfx1250) — matrix operand order repaired and checked by reading, not compiled, not run on hardware
//
#include <hip/hip_runtime.h>
#include <hip/hip_bf16.h>

typedef __attribute__((ext_vector_type(16))) _Float16 v16h;
typedef __attribute__((ext_vector_type(8)))  _Float16 v8h;
typedef __attribute__((ext_vector_type(8)))  float    v8f;
typedef __attribute__((ext_vector_type(4)))  float    v4f;

#define BB 16
#define NN 2048
#define FD 64
#define SCL 0.125f      // 1/sqrt(64)
#define NEGV (-1.0e9f)

__device__ __forceinline__ v8h ld8h(const _Float16* p) { return *(const v8h*)p; }

__device__ __forceinline__ v16h cat16(v8h lo, v8h hi) {
    v16h r;
#pragma unroll
    for (int i = 0; i < 8; ++i) { r[i] = lo[i]; r[i + 8] = hi[i]; }
    return r;
}

// load 8 consecutive f32 and convert to packed f16
__device__ __forceinline__ v8h cvt8(const float* p) {
    v4f a = *(const v4f*)p;
    v4f b = *(const v4f*)(p + 4);
    v8h r;
    r[0] = (_Float16)a[0]; r[1] = (_Float16)a[1]; r[2] = (_Float16)a[2]; r[3] = (_Float16)a[3];
    r[4] = (_Float16)b[0]; r[5] = (_Float16)b[1]; r[6] = (_Float16)b[2]; r[7] = (_Float16)b[3];
    return r;
}

// reduce across the 16 lanes of each half-wave (C/D layout puts a matrix row there)
__device__ __forceinline__ float rmax16(float v) {
    v = fmaxf(v, __shfl_xor(v, 1, 32));
    v = fmaxf(v, __shfl_xor(v, 2, 32));
    v = fmaxf(v, __shfl_xor(v, 4, 32));
    v = fmaxf(v, __shfl_xor(v, 8, 32));
    return v;
}

// ---------------------------------------------------------------------------
// Kernel 0: W[64x64] f32 row-major  ->  Wt[out][k] f16 (transposed), 3 matrices
// ---------------------------------------------------------------------------
__global__ void prep_w(const float* __restrict__ Wq, const float* __restrict__ Wk,
                       const float* __restrict__ Wv, _Float16* __restrict__ Wt) {
    int idx = blockIdx.x * blockDim.x + threadIdx.x;
    if (idx >= 3 * 64 * 64) return;
    int m    = idx >> 12;
    int rem  = idx & 4095;
    int nout = rem >> 6;
    int k    = rem & 63;
    const float* W = (m == 0) ? Wq : (m == 1) ? Wk : Wv;
    Wt[idx] = (_Float16)W[k * 64 + nout];
}

// ---------------------------------------------------------------------------
// Kernel 1: QKV projection with WMMA. One wave per 16-row tile.
// Q,K -> f16 row-major [B][N][64];  V -> f16 transposed [B][64][N]
// ---------------------------------------------------------------------------
__global__ __launch_bounds__(128) void qkv_kernel(
    const float* __restrict__ x, const _Float16* __restrict__ Wt,
    _Float16* __restrict__ Qh, _Float16* __restrict__ Kh, _Float16* __restrict__ Vt) {

    const int lane = threadIdx.x & 31;
    const int wv   = threadIdx.x >> 5;
    const int tile = blockIdx.x * 4 + wv;      // 0..2047
    const int b    = tile >> 7;                // N/16 = 128 tiles per batch
    const int i0   = (tile & 127) << 4;
    const int l    = lane & 15;
    const int h    = lane >> 4;

    // A-fragments of x tile (16x64 split into two 16x32 K-windows), f32 -> f16
    const float* xr = x + ((size_t)b * NN + i0 + l) * FD;
    v16h a0 = cat16(cvt8(xr + h * 16),      cvt8(xr + h * 16 + 8));
    v16h a1 = cat16(cvt8(xr + 32 + h * 16), cvt8(xr + 32 + h * 16 + 8));

#pragma unroll
    for (int m = 0; m < 3; ++m) {
        const _Float16* W = Wt + m * 4096;     // Wt[out][k]
#pragma unroll
        for (int cb = 0; cb < 4; ++cb) {
            const _Float16* wc = W + (size_t)(cb * 16 + l) * 64;   // column (cb*16+l)
            v16h b0 = cat16(ld8h(wc + h * 16),      ld8h(wc + h * 16 + 8));        // k 0..31
            v16h b1 = cat16(ld8h(wc + 32 + h * 16), ld8h(wc + 32 + h * 16 + 8));   // k 32..63
            v8f c = {};
            c = __builtin_amdgcn_wmma_f32_16x16x32_f16(false, a1, false, b1, (short)0, c, false, false);
            c = __builtin_amdgcn_wmma_f32_16x16x32_f16(false, a0, false, b0, (short)0, c, false, false);
            if (m == 2) {
                // V: store transposed; C-layout rows v+8h are consecutive in N
                v8h pk;
#pragma unroll
                for (int v = 0; v < 8; ++v) pk[v] = (_Float16)c[v];
                *(v8h*)(Vt + ((size_t)b * FD + cb * 16 + l) * NN + i0 + 8 * h) = pk;
            } else {
                _Float16* O = (m == 0 ? Qh : Kh) + ((size_t)b * NN + i0) * FD + cb * 16 + l;
#pragma unroll
                for (int v = 0; v < 8; ++v) O[(size_t)(v + 8 * h) * FD] = (_Float16)c[v];
            }
        }
    }
}

// ---------------------------------------------------------------------------
// Kernel 2: flash-style masked attention. One wave per 16-row tile, 4 waves/blk.
// Row-sum of P is computed on the matrix pipe via an all-ones B fragment, so
// the softmax denominator is just a 5th WMMA accumulator (same alpha rescale).
// ---------------------------------------------------------------------------
__global__ __launch_bounds__(128) void attn_kernel(
    const int* __restrict__ adj,
    const _Float16* __restrict__ Qh, const _Float16* __restrict__ Kh,
    const _Float16* __restrict__ Vt, float* __restrict__ out) {

    __shared__ _Float16 plds[4][16 * 32];      // per-wave P tile (16x32 f16)

    const int lane = threadIdx.x & 31;
    const int wv   = threadIdx.x >> 5;
    const int tile = blockIdx.x * 4 + wv;
    const int b    = tile >> 7;
    const int i0   = (tile & 127) << 4;
    const int l    = lane & 15;
    const int h    = lane >> 4;

    // Q A-fragments (two 16x32 K-windows over d=0..63)
    const _Float16* qr = Qh + ((size_t)b * NN + i0 + l) * FD;
    v16h a0 = cat16(ld8h(qr + h * 16),      ld8h(qr + h * 16 + 8));
    v16h a1 = cat16(ld8h(qr + 32 + h * 16), ld8h(qr + 32 + h * 16 + 8));

    // all-ones B fragment: D[m][n] = sum_k A[m][k] for every column n
    v16h ones;
#pragma unroll
    for (int i = 0; i < 16; ++i) ones[i] = (_Float16)1.0f;

    const int*      adjp  = adj + ((size_t)b * NN + i0) * NN;
    const _Float16* kbase = Kh + (size_t)b * NN * FD;
    const _Float16* vbase = Vt + (size_t)b * FD * NN;

    v8f acc0 = {}, acc1 = {}, acc2 = {}, acc3 = {};
    v8f lsum = {};                              // softmax denominator per row
    float mrow[8];
#pragma unroll
    for (int v = 0; v < 8; ++v) mrow[v] = -1e30f;

    _Float16* pw = plds[wv];

    for (int jj = 0; jj < NN; jj += 32) {
        // ---- scores for two 16-column tiles: S = Q · K^T -------------------
        const _Float16* k0 = kbase + (size_t)(jj + l) * FD;
        const _Float16* k1 = kbase + (size_t)(jj + 16 + l) * FD;
        v16h kb00 = cat16(ld8h(k0 + h * 16),      ld8h(k0 + h * 16 + 8));
        v16h kb01 = cat16(ld8h(k0 + 32 + h * 16), ld8h(k0 + 32 + h * 16 + 8));
        v16h kb10 = cat16(ld8h(k1 + h * 16),      ld8h(k1 + h * 16 + 8));
        v16h kb11 = cat16(ld8h(k1 + 32 + h * 16), ld8h(k1 + 32 + h * 16 + 8));

        v8f s0 = {};
        s0 = __builtin_amdgcn_wmma_f32_16x16x32_f16(false, a1, false, kb01, (short)0, s0, false, false);
        s0 = __builtin_amdgcn_wmma_f32_16x16x32_f16(false, a0, false, kb00, (short)0, s0, false, false);
        v8f s1 = {};
        s1 = __builtin_amdgcn_wmma_f32_16x16x32_f16(false, a1, false, kb11, (short)0, s1, false, false);
        s1 = __builtin_amdgcn_wmma_f32_16x16x32_f16(false, a0, false, kb10, (short)0, s1, false, false);

        // ---- mask (non-temporal: adj streams 256MB exactly once) ----------
        float alpha[8];
#pragma unroll
        for (int v = 0; v < 8; ++v) {
            const int* ar = adjp + (size_t)(v + 8 * h) * NN + jj + l;
            int m0 = __builtin_nontemporal_load(ar);
            int m1 = __builtin_nontemporal_load(ar + 16);
            float e0 = s0[v] * SCL; if (m0 == 0) e0 = NEGV;
            float e1 = s1[v] * SCL; if (m1 == 0) e1 = NEGV;
            float tm = rmax16(fmaxf(e0, e1));
            float mn = fmaxf(mrow[v], tm);
            alpha[v] = __expf(mrow[v] - mn);
            mrow[v]  = mn;
            float p0 = __expf(e0 - mn);
            float p1 = __expf(e1 - mn);
            pw[(v + 8 * h) * 32 + l]      = (_Float16)p0;
            pw[(v + 8 * h) * 32 + 16 + l] = (_Float16)p1;
        }

        // read P back as an A-fragment (wave-private LDS, DS ops are in-order)
        v16h pa = cat16(ld8h(pw + l * 32 + 16 * h), ld8h(pw + l * 32 + 16 * h + 8));

        // rescale all accumulators (output + denominator) by alpha per row
#pragma unroll
        for (int v = 0; v < 8; ++v) {
            acc0[v] *= alpha[v]; acc1[v] *= alpha[v];
            acc2[v] *= alpha[v]; acc3[v] *= alpha[v];
            lsum[v] *= alpha[v];
        }

        // ---- denominator: lsum += P · 1  (rides the XDL pipe) -------------
        lsum = __builtin_amdgcn_wmma_f32_16x16x32_f16(false, pa, false, ones, (short)0, lsum, false, false);

        // ---- O += P(16x32) · V(32x64), V pre-transposed so loads are linear
        v16h vb;
        vb = cat16(ld8h(vbase + (size_t)(l)      * NN + jj + h * 16),
                   ld8h(vbase + (size_t)(l)      * NN + jj + h * 16 + 8));
        acc0 = __builtin_amdgcn_wmma_f32_16x16x32_f16(false, pa, false, vb, (short)0, acc0, false, false);
        vb = cat16(ld8h(vbase + (size_t)(16 + l) * NN + jj + h * 16),
                   ld8h(vbase + (size_t)(16 + l) * NN + jj + h * 16 + 8));
        acc1 = __builtin_amdgcn_wmma_f32_16x16x32_f16(false, pa, false, vb, (short)0, acc1, false, false);
        vb = cat16(ld8h(vbase + (size_t)(32 + l) * NN + jj + h * 16),
                   ld8h(vbase + (size_t)(32 + l) * NN + jj + h * 16 + 8));
        acc2 = __builtin_amdgcn_wmma_f32_16x16x32_f16(false, pa, false, vb, (short)0, acc2, false, false);
        vb = cat16(ld8h(vbase + (size_t)(48 + l) * NN + jj + h * 16),
                   ld8h(vbase + (size_t)(48 + l) * NN + jj + h * 16 + 8));
        acc3 = __builtin_amdgcn_wmma_f32_16x16x32_f16(false, pa, false, vb, (short)0, acc3, false, false);
    }

    // ---- normalize and store f32 output -----------------------------------
    float* orow = out + ((size_t)b * NN + i0) * FD + l;
#pragma unroll
    for (int v = 0; v < 8; ++v) {
        float inv = 1.0f / lsum[v];
        float* orr = orow + (size_t)(v + 8 * h) * FD;
        orr[0]  = acc0[v] * inv;
        orr[16] = acc1[v] * inv;
        orr[32] = acc2[v] * inv;
        orr[48] = acc3[v] * inv;
    }
}

// ---------------------------------------------------------------------------
extern "C" void kernel_launch(void* const* d_in, const int* in_sizes, int n_in,
                              void* d_out, int out_size, void* d_ws, size_t ws_size,
                              hipStream_t stream) {
    (void)in_sizes; (void)n_in; (void)out_size; (void)ws_size;
    const float* x   = (const float*)d_in[0];
    const int*   adj = (const int*)d_in[1];
    const float* Wq  = (const float*)d_in[2];
    const float* Wk  = (const float*)d_in[3];
    const float* Wv  = (const float*)d_in[4];
    float*       out = (float*)d_out;

    _Float16* Wt = (_Float16*)d_ws;                       // 3*64*64 halfs
    _Float16* Qh = Wt + 3 * 4096;                         // B*N*64 halfs
    _Float16* Kh = Qh + (size_t)BB * NN * FD;
    _Float16* Vt = Kh + (size_t)BB * NN * FD;             // transposed [B][64][N]

    prep_w<<<48, 256, 0, stream>>>(Wq, Wk, Wv, Wt);
    qkv_kernel<<<512, 128, 0, stream>>>(x, Wt, Qh, Kh, Vt);
    attn_kernel<<<512, 128, 0, stream>>>(adj, Qh, Kh, Vt, out);
}
